// LFADS_59124519797090
// MI455X (gfx1250) — matrix operand order repaired and checked by reading, not compiled, not run on hardware
//
#include <hip/hip_runtime.h>
#include <hip/hip_bf16.h>
#include <math.h>

// ---------------- problem constants ----------------
#define Bc    128
#define Tc    1000
#define DINc  64
#define Ec    256
#define Fc    64
#define Uc    2
#define H1c   80
#define DOUTc 64
#define G3E   768          // 3*E
#define CLIPc 5.0f
#define LVMINc (-9.210340371976182f)   // log(1e-4)

// ---------------- persistent grid ----------------
#define NWG    32
#define NTHR   256
#define NLANES (NWG * NTHR)     // 8192 threads
#define NWAVES (NLANES / 32)    // 256 waves

typedef __bf16 bf16_t;
typedef __attribute__((ext_vector_type(16))) __bf16 bf16x16;
typedef __attribute__((ext_vector_type(8)))  __bf16 bf16x8;
typedef __attribute__((ext_vector_type(8)))  float  f32x8;

// global (addrspace 1) qualified views: force global_load/_store instead of flat
#define GAS __attribute__((address_space(1)))
typedef GAS const bf16x8  gv_bf16x8;
typedef GAS const bf16x16 gv_bf16x16;
typedef unsigned long long uptr_t;

__device__ __forceinline__ GAS const bf16_t* gload_cbf(const bf16_t* p) {
  return (GAS const bf16_t*)(uptr_t)p;
}
__device__ __forceinline__ GAS bf16_t* gptr_bf(bf16_t* p) {
  return (GAS bf16_t*)(uptr_t)p;
}
__device__ __forceinline__ GAS float* gptr_f(float* p) {
  return (GAS float*)(uptr_t)p;
}
__device__ __forceinline__ GAS const float* gptr_cf(const float* p) {
  return (GAS const float*)(uptr_t)p;
}

// ============================================================
// WMMA tile helpers (bf16 A/B, f32 accum, 16x16x32)
//  A: row-major activations (M x K), per-lane frag = two 16B global loads
//  B: weights row-major (N x K)  => computing X @ W^T; per-lane frag
//     = two 8-bf16 runs of one weight row, same k order as A (two 16B global loads)
// ============================================================
__device__ __forceinline__ f32x8 gemm_k(f32x8 acc, const bf16_t* A, int lda,
                                        const bf16_t* W, int ldw, int K, int lane) {
  const int half = lane >> 4;
  const int mn   = lane & 15;
  const uptr_t ap = (uptr_t)(A + (size_t)mn * lda + half * 8);
  const uptr_t wp = (uptr_t)(W + (size_t)mn * ldw + half * 8);
  __builtin_prefetch((const void*)wp, 0, 1);   // global_prefetch_b8
  for (int k = 0; k < K; k += 32) {
    bf16x16 a, b;
    *(bf16x8*)&a       = *(const gv_bf16x8*)(ap + 2u * (unsigned)k);        // K = half*8 + [0..7]
    *((bf16x8*)&a + 1) = *(const gv_bf16x8*)(ap + 2u * (unsigned)k + 32u);  // K = 16 + half*8 + [0..7]
    *(bf16x8*)&b       = *(const gv_bf16x8*)(wp + 2u * (unsigned)k);        // K = half*8 + [0..7]
    *((bf16x8*)&b + 1) = *(const gv_bf16x8*)(wp + 2u * (unsigned)k + 32u);  // K = 16 + half*8 + [0..7]
    acc = __builtin_amdgcn_wmma_f32_16x16x32_bf16(false, a, false, b,
                                                  (short)0, acc, false, false);
  }
  return acc;
}

// C/D layout: lane n = lane&15, mhalf = lane>>4, VGPR r -> row mhalf*8+r
__device__ __forceinline__ void store_tile_f32(float* OUT, int ld, int mt, int nt,
                                               f32x8 acc, const float* bias, int lane) {
  const int n = lane & 15, mh = lane >> 4;
  const float bv = bias ? gptr_cf(bias)[nt * 16 + n] : 0.f;
  GAS float* op = gptr_f(OUT + (size_t)(mt * 16) * ld + nt * 16 + n);
#pragma unroll
  for (int r = 0; r < 8; ++r) op[(size_t)(mh * 8 + r) * ld] = acc[r] + bv;
}

__device__ __forceinline__ void store_tile_bf16(bf16_t* OUT, int ld, int mt, int nt,
                                                f32x8 acc, const float* bias,
                                                bool relu, int lane) {
  const int n = lane & 15, mh = lane >> 4;
  const float bv = bias ? gptr_cf(bias)[nt * 16 + n] : 0.f;
  GAS bf16_t* op = gptr_bf(OUT + (size_t)(mt * 16) * ld + nt * 16 + n);
#pragma unroll
  for (int r = 0; r < 8; ++r) {
    float v = acc[r] + bv;
    if (relu) v = fmaxf(v, 0.f);
    op[(size_t)(mh * 8 + r) * ld] = (bf16_t)v;
  }
}

// ---------------- global spin barrier (monotonic counter) ----------------
__device__ __forceinline__ void gbar(unsigned* bar, unsigned& gen) {
  __syncthreads();
  gen += NWG;
  if (threadIdx.x == 0) {
    __hip_atomic_fetch_add(bar, 1u, __ATOMIC_ACQ_REL, __HIP_MEMORY_SCOPE_AGENT);
    while (__hip_atomic_load(bar, __ATOMIC_ACQUIRE, __HIP_MEMORY_SCOPE_AGENT) < gen)
      __builtin_amdgcn_s_sleep(1);
  }
  __syncthreads();
}

__device__ __forceinline__ float sigf(float x) { return 1.f / (1.f + __expf(-x)); }

// GRU gate update: h' = min((1-z)*n + z*h, CLIP); writes fp32 + bf16 (+states)
__device__ __forceinline__ void gru_update(const float* gi, const float* gh, float* h,
                                           bf16_t* hbf, bf16_t* st, int gtid) {
  GAS float* hg = gptr_f(h);
  GAS bf16_t* hbg = gptr_bf(hbf);
  GAS bf16_t* stg = st ? gptr_bf(st) : (GAS bf16_t*)0;
  for (int i = gtid; i < Bc * Ec; i += NLANES) {
    const int m = i >> 8, j = i & 255;
    GAS const float* gim = gptr_cf(gi) + (size_t)m * G3E;
    GAS const float* ghm = gptr_cf(gh) + (size_t)m * G3E;
    float r = sigf(gim[j] + ghm[j]);
    float z = sigf(gim[j + Ec] + ghm[j + Ec]);
    float n = tanhf(gim[j + 2 * Ec] + r * ghm[j + 2 * Ec]);
    float hv = fminf((1.f - z) * n + z * hg[i], CLIPc);
    hg[i] = hv;
    bf16_t hb = (bf16_t)hv;
    hbg[i] = hb;
    if (stg) stg[i] = hb;
  }
}

// ---------------- parameter block ----------------
struct LfadsP {
  // fp32 originals needed at runtime
  const float* eps_g0;
  const float* eps_u;
  const float* enc_bih[4];  // q: 0=egf 1=ecf 2=egb 3=ecb
  const float* enc_bhh[4];
  const float* con_bih; const float* con_bhh;
  const float* gen_wih; const float* gen_bih; const float* gen_bhh;
  const float* g0m_b;  const float* g0v_b;
  const float* um_w; const float* um_b; const float* uv_w; const float* uv_b;
  const float* fac_b; const float* f1_b; const float* cl_b;
  // packed bf16
  const bf16_t* xbf;                         // (B, T, DIN)
  const bf16_t* enc_wih_bf[4];               // (768, 64)
  const bf16_t* enc_whh_bf[4];               // (768, 256)
  const bf16_t* con_wih_bf;                  // (768, 576)
  const bf16_t* con_whh_bf;                  // (768, 256)
  const bf16_t* gen_whh_bf;                  // (768, 256)
  const bf16_t* g0m_bf; const bf16_t* g0v_bf;// (256, 512)
  const bf16_t* fac_bf;                      // (64, 256)
  const bf16_t* f1_bf;                       // (80, 64)
  const bf16_t* cl_bf;                       // (64, 96)  zero-padded K
  // state / scratch
  float*  h32;        // 6 * B*E  (hgf, hcf, hgb, hcb, c, g)
  bf16_t* hbf;        // 6 * B*E
  bf16_t* ef;         // (T, B, E)
  bf16_t* eb;         // (T, B, E)
  float*  gi_s;       // 4 * B*768  ([0] reused as con GI)
  float*  gh_s;       // 4 * B*768  ([0] con GH, [1] gen GH)
  float*  g0mean; float* g0lv;   // B*E each
  float*  u;          // B*U
  bf16_t* fbf;        // B*64
  bf16_t* ffbf;       // B*96 (cols 80..95 stay zero)
  unsigned* bar;
  float*  out;        // (B, T, DOUT)
};

// ============================================================
// setup kernels
// ============================================================
__global__ void pack_bf16_k(bf16_t* dst, const float* src, int n) {
  int i = blockIdx.x * blockDim.x + threadIdx.x;
  if (i < n) gptr_bf(dst)[i] = (bf16_t)gptr_cf(src)[i];
}

__global__ void pack_cl_pad_k(bf16_t* dst, const float* src) {  // (64,80)->(64,96)
  int i = blockIdx.x * blockDim.x + threadIdx.x;
  if (i < DOUTc * 96) {
    int r = i / 96, c = i % 96;
    gptr_bf(dst)[i] = (c < H1c) ? (bf16_t)gptr_cf(src)[r * H1c + c] : (bf16_t)0.f;
  }
}

__global__ void init_ws_k(float* h32, bf16_t* hbf, bf16_t* ffbf, unsigned* bar) {
  int i = blockIdx.x * blockDim.x + threadIdx.x;
  if (i < 6 * Bc * Ec) { gptr_f(h32)[i] = 0.f; gptr_bf(hbf)[i] = (bf16_t)0.f; }
  if (i < Bc * 96) gptr_bf(ffbf)[i] = (bf16_t)0.f;
  if (i == 0) bar[0] = 0u;
}

// ============================================================
// persistent cooperative kernel: whole LFADS forward
// ============================================================
__global__ __launch_bounds__(NTHR) void lfads_persist(LfadsP p) {
  const int tid  = threadIdx.x;
  const int gtid = blockIdx.x * NTHR + tid;
  const int lane = tid & 31;
  const int wv   = gtid >> 5;
  unsigned gen = 0;

  float*  h32[6]; bf16_t* hbf[6];
  for (int q = 0; q < 6; ++q) { h32[q] = p.h32 + q * (Bc * Ec); hbf[q] = p.hbf + q * (Bc * Ec); }

  // ================= encoder scans (fwd + bwd fused per step) =================
  for (int t = 0; t < Tc; ++t) {
    // p1: 8 GEMMs: GI (K=64) and GH (K=256) for q in {egf,ecf,egb,ecb}
    for (int job = wv; job < 8 * 384; job += NWAVES) {
      const int mat = job / 384, tl = job % 384;
      const int mt = tl / 48, nt = tl % 48;
      f32x8 acc = {};
      if (mat < 4) {
        const int q  = mat;
        const int tt = (q >= 2) ? (Tc - 1 - t) : t;
        const bf16_t* A = p.xbf + (size_t)tt * DINc + (size_t)(mt * 16) * (Tc * DINc);
        acc = gemm_k(acc, A, Tc * DINc, p.enc_wih_bf[q] + (size_t)(nt * 16) * DINc,
                     DINc, DINc, lane);
        store_tile_f32(p.gi_s + (size_t)q * (Bc * G3E), G3E, mt, nt, acc,
                       p.enc_bih[q], lane);
      } else {
        const int q = mat - 4;
        acc = gemm_k(acc, hbf[q] + (size_t)(mt * 16) * Ec, Ec,
                     p.enc_whh_bf[q] + (size_t)(nt * 16) * Ec, Ec, Ec, lane);
        store_tile_f32(p.gh_s + (size_t)q * (Bc * G3E), G3E, mt, nt, acc,
                       p.enc_bhh[q], lane);
      }
    }
    gbar(p.bar, gen);
    // p2: gate updates; hc states recorded as bf16 for the generator
    for (int q = 0; q < 4; ++q) {
      bf16_t* st = nullptr;
      if (q == 1) st = p.ef + (size_t)t * Bc * Ec;
      if (q == 3) st = p.eb + (size_t)(Tc - 1 - t) * Bc * Ec;
      gru_update(p.gi_s + (size_t)q * (Bc * G3E), p.gh_s + (size_t)q * (Bc * G3E),
                 h32[q], hbf[q], st, gtid);
    }
    gbar(p.bar, gen);
  }

  // ================= g0 sample =================
  // pA: mean / logvar GEMMs, K=512 split as [hg_f | hg_b]
  for (int job = wv; job < 2 * 8 * 16; job += NWAVES) {
    const int which = job / 128, tl = job % 128;
    const int mt = tl / 16, nt = tl % 16;
    const bf16_t* W = which ? p.g0v_bf : p.g0m_bf;
    f32x8 acc = {};
    acc = gemm_k(acc, hbf[0] + (size_t)(mt * 16) * Ec, Ec,
                 W + (size_t)(nt * 16) * 512, 512, Ec, lane);
    acc = gemm_k(acc, hbf[2] + (size_t)(mt * 16) * Ec, Ec,
                 W + (size_t)(nt * 16) * 512 + Ec, 512, Ec, lane);
    store_tile_f32(which ? p.g0lv : p.g0mean, Ec, mt, nt, acc,
                   which ? p.g0v_b : p.g0m_b, lane);
  }
  gbar(p.bar, gen);
  // pB: g = eps * exp(0.5*max(lv,LVMIN)) + mean  -> h[5]
  for (int i = gtid; i < Bc * Ec; i += NLANES) {
    float lv = fmaxf(gptr_cf(p.g0lv)[i], LVMINc);
    float g  = gptr_cf(p.eps_g0)[i] * __expf(0.5f * lv) + gptr_cf(p.g0mean)[i];
    gptr_f(h32[5])[i] = g; gptr_bf(hbf[5])[i] = (bf16_t)g;
  }
  gbar(p.bar, gen);
  // pC: f0 = g @ fac^T
  for (int job = wv; job < 32; job += NWAVES) {
    const int mt = job / 4, nt = job % 4;
    f32x8 acc = {};
    acc = gemm_k(acc, hbf[5] + (size_t)(mt * 16) * Ec, Ec,
                 p.fac_bf + (size_t)(nt * 16) * Ec, Ec, Ec, lane);
    store_tile_bf16(p.fbf, Fc, mt, nt, acc, p.fac_b, false, lane);
  }
  gbar(p.bar, gen);

  // ================= generator scan =================
  float* gi_con = p.gi_s;
  float* gh_con = p.gh_s;
  float* gh_gen = p.gh_s + (size_t)(Bc * G3E);
  for (int t = 0; t < Tc; ++t) {
    // p1: con GI (K = 256|256|64 segments), con GH (A=c), gen GH (A=g)
    for (int job = wv; job < 3 * 384; job += NWAVES) {
      const int mat = job / 384, tl = job % 384;
      const int mt = tl / 48, nt = tl % 48;
      f32x8 acc = {};
      if (mat == 0) {
        const bf16_t* W = p.con_wih_bf + (size_t)(nt * 16) * 576;
        acc = gemm_k(acc, p.ef + (size_t)t * Bc * Ec + (size_t)(mt * 16) * Ec, Ec, W,       576, Ec, lane);
        acc = gemm_k(acc, p.eb + (size_t)t * Bc * Ec + (size_t)(mt * 16) * Ec, Ec, W + 256, 576, Ec, lane);
        acc = gemm_k(acc, p.fbf + (size_t)(mt * 16) * Fc, Fc,                      W + 512, 576, Fc, lane);
        store_tile_f32(gi_con, G3E, mt, nt, acc, p.con_bih, lane);
      } else if (mat == 1) {
        acc = gemm_k(acc, hbf[4] + (size_t)(mt * 16) * Ec, Ec,
                     p.con_whh_bf + (size_t)(nt * 16) * Ec, Ec, Ec, lane);
        store_tile_f32(gh_con, G3E, mt, nt, acc, p.con_bhh, lane);
      } else {
        acc = gemm_k(acc, hbf[5] + (size_t)(mt * 16) * Ec, Ec,
                     p.gen_whh_bf + (size_t)(nt * 16) * Ec, Ec, Ec, lane);
        store_tile_f32(gh_gen, G3E, mt, nt, acc, p.gen_bhh, lane);
      }
    }
    gbar(p.bar, gen);
    // p2: controller GRU update -> c
    gru_update(gi_con, gh_con, h32[4], hbf[4], nullptr, gtid);
    gbar(p.bar, gen);
    // p3: u sample (N=2, VALU dots over c rows)
    for (int i = gtid; i < Bc * Uc; i += NLANES) {
      const int m = i / Uc, d = i % Uc;
      GAS const float* c   = gptr_cf(h32[4]) + (size_t)m * Ec;
      GAS const float* wm  = gptr_cf(p.um_w) + (size_t)d * Ec;
      GAS const float* wv2 = gptr_cf(p.uv_w) + (size_t)d * Ec;
      float sm = p.um_b[d], sv = p.uv_b[d];
      for (int k = 0; k < Ec; ++k) { sm += c[k] * wm[k]; sv += c[k] * wv2[k]; }
      float lv = fmaxf(sv, LVMINc);
      gptr_f(p.u)[i] = gptr_cf(p.eps_u)[((size_t)m * Tc + t) * Uc + d] * __expf(0.5f * lv) + sm;
    }
    gbar(p.bar, gen);
    // p4: generator GRU update; K=2 input projection inline (VALU)
    for (int i = gtid; i < Bc * Ec; i += NLANES) {
      const int m = i >> 8, j = i & 255;
      const float u0 = gptr_cf(p.u)[m * Uc], u1 = gptr_cf(p.u)[m * Uc + 1];
      GAS const float* gh = gptr_cf(gh_gen) + (size_t)m * G3E;
      GAS const float* wih = gptr_cf(p.gen_wih);
      GAS const float* bih = gptr_cf(p.gen_bih);
      float gir = u0 * wih[(j) * Uc]       + u1 * wih[(j) * Uc + 1]       + bih[j];
      float giz = u0 * wih[(j + 256) * Uc] + u1 * wih[(j + 256) * Uc + 1] + bih[j + 256];
      float gin = u0 * wih[(j + 512) * Uc] + u1 * wih[(j + 512) * Uc + 1] + bih[j + 512];
      float r = sigf(gir + gh[j]);
      float z = sigf(giz + gh[j + 256]);
      float n = tanhf(gin + r * gh[j + 512]);
      float g = fminf((1.f - z) * n + z * gptr_f(h32[5])[i], CLIPc);
      gptr_f(h32[5])[i] = g; gptr_bf(hbf[5])[i] = (bf16_t)g;
    }
    gbar(p.bar, gen);
    // p5: f = g @ fac^T
    for (int job = wv; job < 32; job += NWAVES) {
      const int mt = job / 4, nt = job % 4;
      f32x8 acc = {};
      acc = gemm_k(acc, hbf[5] + (size_t)(mt * 16) * Ec, Ec,
                   p.fac_bf + (size_t)(nt * 16) * Ec, Ec, Ec, lane);
      store_tile_bf16(p.fbf, Fc, mt, nt, acc, p.fac_b, false, lane);
    }
    gbar(p.bar, gen);
    // p6: ff = relu(f @ f1^T)  (N=80 -> 5 tiles), stored K-padded to 96
    for (int job = wv; job < 40; job += NWAVES) {
      const int mt = job / 5, nt = job % 5;
      f32x8 acc = {};
      acc = gemm_k(acc, p.fbf + (size_t)(mt * 16) * Fc, Fc,
                   p.f1_bf + (size_t)(nt * 16) * Fc, Fc, Fc, lane);
      store_tile_bf16(p.ffbf, 96, mt, nt, acc, p.f1_b, true, lane);
    }
    gbar(p.bar, gen);
    // p7: out = ff @ cl^T (K padded to 96) -> d_out[(m*T + t)*64 + n]
    for (int job = wv; job < 32; job += NWAVES) {
      const int mt = job / 4, nt = job % 4;
      f32x8 acc = {};
      acc = gemm_k(acc, p.ffbf + (size_t)(mt * 16) * 96, 96,
                   p.cl_bf + (size_t)(nt * 16) * 96, 96, 96, lane);
      store_tile_f32(p.out + (size_t)t * DOUTc, Tc * DOUTc, mt, nt, acc, p.cl_b, lane);
    }
    // no barrier needed here: next p1 touches disjoint buffers, first reuse is
    // barrier-separated by p1->p2
  }
}

// ============================================================
// host launcher
// ============================================================
extern "C" void kernel_launch(void* const* d_in, const int* in_sizes, int n_in,
                              void* d_out, int out_size, void* d_ws, size_t ws_size,
                              hipStream_t stream) {
  (void)in_sizes; (void)n_in; (void)out_size;
  const float* in[41];
  for (int i = 0; i < 41; ++i) in[i] = (const float*)d_in[i];
  // input index map (setup_inputs order)
  // 0 x, 1 eps_g0, 2 eps_u, 3..6 egf, 7..10 egb, 11..14 ecf, 15..18 ecb,
  // 19..22 con, 23..26 gen, 27/28 g0m, 29/30 g0v, 31/32 um, 33/34 uv,
  // 35/36 fac, 37/38 f1, 39/40 cl
  const int QW[4] = {3, 11, 7, 15};   // internal q order: egf, ecf, egb, ecb

  char* base = (char*)d_ws;
  size_t off = 0;
  auto take = [&](size_t elems, size_t esz) -> size_t {
    off = (off + 255) & ~(size_t)255;
    size_t o = off; off += elems * esz; return o;
  };

  size_t o_bar    = take(64, 4);
  size_t o_xbf    = take((size_t)Bc * Tc * DINc, 2);
  size_t o_ewih[4], o_ewhh[4];
  for (int q = 0; q < 4; ++q) { o_ewih[q] = take(G3E * DINc, 2); o_ewhh[q] = take(G3E * Ec, 2); }
  size_t o_cwih   = take(G3E * 576, 2);
  size_t o_cwhh   = take(G3E * Ec, 2);
  size_t o_gwhh   = take(G3E * Ec, 2);
  size_t o_g0m    = take(Ec * 512, 2);
  size_t o_g0v    = take(Ec * 512, 2);
  size_t o_fac    = take(Fc * Ec, 2);
  size_t o_f1     = take(H1c * Fc, 2);
  size_t o_cl     = take(DOUTc * 96, 2);
  size_t o_h32    = take(6 * (size_t)Bc * Ec, 4);
  size_t o_hbf    = take(6 * (size_t)Bc * Ec, 2);
  size_t o_ef     = take((size_t)Tc * Bc * Ec, 2);
  size_t o_eb     = take((size_t)Tc * Bc * Ec, 2);
  size_t o_gis    = take(4 * (size_t)Bc * G3E, 4);
  size_t o_ghs    = take(4 * (size_t)Bc * G3E, 4);
  size_t o_g0mean = take((size_t)Bc * Ec, 4);
  size_t o_g0lv   = take((size_t)Bc * Ec, 4);
  size_t o_u      = take(Bc * Uc, 4);
  size_t o_fbf    = take((size_t)Bc * Fc, 2);
  size_t o_ffbf   = take((size_t)Bc * 96, 2);
  if (off > ws_size) return;  // workspace too small; bail deterministically

  auto bfp = [&](size_t o) { return (bf16_t*)(base + o); };
  auto f32p = [&](size_t o) { return (float*)(base + o); };

  // ---- pack / init ----
  auto pk = [&](bf16_t* dst, const float* src, int n) {
    pack_bf16_k<<<(n + 255) / 256, 256, 0, stream>>>(dst, src, n);
  };
  pk(bfp(o_xbf), in[0], Bc * Tc * DINc);
  for (int q = 0; q < 4; ++q) {
    pk(bfp(o_ewih[q]), in[QW[q]],     G3E * DINc);
    pk(bfp(o_ewhh[q]), in[QW[q] + 1], G3E * Ec);
  }
  pk(bfp(o_cwih), in[19], G3E * 576);
  pk(bfp(o_cwhh), in[20], G3E * Ec);
  pk(bfp(o_gwhh), in[24], G3E * Ec);
  pk(bfp(o_g0m),  in[27], Ec * 512);
  pk(bfp(o_g0v),  in[29], Ec * 512);
  pk(bfp(o_fac),  in[35], Fc * Ec);
  pk(bfp(o_f1),   in[37], H1c * Fc);
  pack_cl_pad_k<<<(DOUTc * 96 + 255) / 256, 256, 0, stream>>>(bfp(o_cl), in[39]);
  init_ws_k<<<(6 * Bc * Ec + 255) / 256, 256, 0, stream>>>(
      f32p(o_h32), bfp(o_hbf), bfp(o_ffbf), (unsigned*)(base + o_bar));

  // ---- params ----
  LfadsP p;
  p.eps_g0 = in[1]; p.eps_u = in[2];
  for (int q = 0; q < 4; ++q) { p.enc_bih[q] = in[QW[q] + 2]; p.enc_bhh[q] = in[QW[q] + 3]; }
  p.con_bih = in[21]; p.con_bhh = in[22];
  p.gen_wih = in[23]; p.gen_bih = in[25]; p.gen_bhh = in[26];
  p.g0m_b = in[28]; p.g0v_b = in[30];
  p.um_w = in[31]; p.um_b = in[32]; p.uv_w = in[33]; p.uv_b = in[34];
  p.fac_b = in[36]; p.f1_b = in[38]; p.cl_b = in[40];
  p.xbf = bfp(o_xbf);
  for (int q = 0; q < 4; ++q) { p.enc_wih_bf[q] = bfp(o_ewih[q]); p.enc_whh_bf[q] = bfp(o_ewhh[q]); }
  p.con_wih_bf = bfp(o_cwih); p.con_whh_bf = bfp(o_cwhh); p.gen_whh_bf = bfp(o_gwhh);
  p.g0m_bf = bfp(o_g0m); p.g0v_bf = bfp(o_g0v); p.fac_bf = bfp(o_fac);
  p.f1_bf = bfp(o_f1); p.cl_bf = bfp(o_cl);
  p.h32 = f32p(o_h32); p.hbf = bfp(o_hbf);
  p.ef = bfp(o_ef); p.eb = bfp(o_eb);
  p.gi_s = f32p(o_gis); p.gh_s = f32p(o_ghs);
  p.g0mean = f32p(o_g0mean); p.g0lv = f32p(o_g0lv);
  p.u = f32p(o_u); p.fbf = bfp(o_fbf); p.ffbf = bfp(o_ffbf);
  p.bar = (unsigned*)(base + o_bar);
  p.out = (float*)d_out;

  lfads_persist<<<NWG, NTHR, 0, stream>>>(p);
}
